// ptBEVnet_70317204570752
// MI455X (gfx1250) — matrix operand order repaired and checked by reading, not compiled, not run on hardware
//
#include <hip/hip_runtime.h>

// ---------------- constants ----------------
#define GXX 480
#define GYY 360
#define NUM_VOX (GXX * GYY)      // 172800
#define NPTS 100000
#define BATCH 2
#define MAXPT 256
#define BN_EPS 1e-5f

// tiles
#define TILE_M 64
#define TILES_PER_B ((NPTS + TILE_M - 1) / TILE_M)   // 1563
#define VOX_TILES (NUM_VOX / TILE_M)                 // 2700 (exact)

// pitches (bytes) for bf16 activation rows in LDS: K*2 + 16  (pitch/16 odd -> conflict free b128)
#define P32  80
#define P64  144
#define P128 272
#define P256 528
#define P512 1040

// fragment counts (each fragment = 32 lanes x 8 dwords = 1024 bytes)
#define NFRAG0 4      // 64 out,  K32 pad : NT=4  KT=1
#define NFRAG1 16     // 128 out, K64     : NT=8  KT=2
#define NFRAG2 64     // 256 out, K128    : NT=16 KT=4
#define NFRAG3 256    // 512 out, K256    : NT=32 KT=8
#define NFRAGC 32     // 32 out,  K512    : NT=2  KT=16

// ---------------- workspace layout (bytes) ----------------
constexpr size_t VMAX_BYTES = (size_t)BATCH * NUM_VOX * 512 * 4;  // 707,788,800
constexpr size_t CNT_OFF    = VMAX_BYTES;
constexpr size_t CNT_BYTES  = (size_t)BATCH * NUM_VOX * 4;        // 1,382,400
constexpr size_t PVOX_OFF   = CNT_OFF + CNT_BYTES;
constexpr size_t PVOX_BYTES = (size_t)BATCH * NPTS * 4;
constexpr size_t WF0_OFF    = PVOX_OFF + PVOX_BYTES;
constexpr size_t WF1_OFF    = WF0_OFF + (size_t)NFRAG0 * 1024;
constexpr size_t WF2_OFF    = WF1_OFF + (size_t)NFRAG1 * 1024;
constexpr size_t WF3_OFF    = WF2_OFF + (size_t)NFRAG2 * 1024;
constexpr size_t WFC_OFF    = WF3_OFF + (size_t)NFRAG3 * 1024;
constexpr size_t BIAS_OFF   = WFC_OFF + (size_t)NFRAGC * 1024;    // 992 floats
constexpr size_t ZERO_U4    = (VMAX_BYTES + CNT_BYTES) / 16;      // uint4 count

// ---------------- types ----------------
typedef __attribute__((ext_vector_type(16))) __bf16    v16bf;
typedef __attribute__((ext_vector_type(2)))  __bf16    v2bf;
typedef __attribute__((ext_vector_type(8)))  float     v8f;
typedef __attribute__((ext_vector_type(8)))  unsigned  v8u;

// ---------------- helpers ----------------
// native f32 <-> bf16 conversion (backend lowers to v_cvt_pk_bf16_f32 on gfx1250)
__device__ __forceinline__ unsigned short f2bf(float f) {
    __bf16 h = (__bf16)f;
    return __builtin_bit_cast(unsigned short, h);
}
__device__ __forceinline__ float bf2f(unsigned short h) {
    return (float)__builtin_bit_cast(__bf16, h);
}
// order-preserving float -> u32 map (for atomicMax)
__device__ __forceinline__ unsigned encf(float f) {
    unsigned b = __float_as_uint(f);
    return (b & 0x80000000u) ? ~b : (b | 0x80000000u);
}
__device__ __forceinline__ float decf(unsigned u) {
    unsigned b = (u & 0x80000000u) ? (u ^ 0x80000000u) : ~u;
    return __uint_as_float(b);
}

// A fragment (16x32 bf16) from row-major pitched LDS buffer.
// ISA layout: lanes 0-15 row M=lane, VGPR0-3 K=0..7, VGPR4-7 K=16..23;
// lanes 16-31 same rows, K=8..15 and 24..31  -> two contiguous 16B spans per lane.
__device__ __forceinline__ v16bf load_a_frag(const unsigned char* buf, int pitch,
                                             int mt, int kt, int hi, int ln) {
    const unsigned char* rp = buf + (size_t)(mt * 16 + ln) * pitch + kt * 64 + hi * 16;
    uint4 x0 = *(const uint4*)rp;
    uint4 x1 = *(const uint4*)(rp + 32);
    v8u av = {x0.x, x0.y, x0.z, x0.w, x1.x, x1.y, x1.z, x1.w};
    return __builtin_bit_cast(v16bf, av);
}

// B fragment from pre-swizzled global buffer: [frag][p][lane] dwords, fully coalesced.
__device__ __forceinline__ v16bf load_b_frag(const unsigned* wf, int frag, int lane) {
    v8u u;
#pragma unroll
    for (int p = 0; p < 8; ++p) u[p] = wf[((size_t)frag * 8 + p) * 32 + lane];
    return __builtin_bit_cast(v16bf, u);
}

// C-fragment epilogue: packed f32->bf16 conversion (v_cvt_pk_bf16_f32) per row-pair,
// ReLU in packed bf16 domain (v_pk_max_num_bf16), two b16 LDS stores (pitch apart).
template <bool RELU>
__device__ __forceinline__ void store_c(unsigned char* __restrict__ outBuf, int pitchOut,
                                        int mt, int ntile, const v8f& acc, int hi, int ln) {
#pragma unroll
    for (int r = 0; r < 8; r += 2) {
        v2bf h = {(__bf16)acc[r], (__bf16)acc[r + 1]};
        if (RELU) {
            v2bf z = {(__bf16)0.0f, (__bf16)0.0f};
            h = __builtin_elementwise_max(h, z);
        }
        unsigned u = __builtin_bit_cast(unsigned, h);
        unsigned char* base = outBuf + (size_t)(mt * 16 + r + hi * 8) * pitchOut +
                              (ntile * 16 + ln) * 2;
        *(unsigned short*)base = (unsigned short)u;
        *(unsigned short*)(base + pitchOut) = (unsigned short)(u >> 16);
    }
}

// one MLP layer: [64 x (KT*32)] @ W^T -> [64 x (NTW*4waves*16)], bf16 in/out in LDS
template <int KT, int NTW, bool RELU>
__device__ __forceinline__ void mlp_layer(const unsigned* __restrict__ wf,
                                          const float* __restrict__ bias,
                                          const unsigned char* __restrict__ inBuf, int pitchIn,
                                          unsigned char* __restrict__ outBuf, int pitchOut,
                                          int wave, int lane) {
    const int hi = lane >> 4, ln = lane & 15;
    for (int i = 0; i < NTW; ++i) {
        const int ntile = wave * NTW + i;
        v16bf Bf[KT];
#pragma unroll
        for (int kt = 0; kt < KT; ++kt) Bf[kt] = load_b_frag(wf, ntile * KT + kt, lane);
        const float bv = bias[ntile * 16 + ln];
#pragma unroll
        for (int mt = 0; mt < 4; ++mt) {
            v8f acc = {bv, bv, bv, bv, bv, bv, bv, bv};
#pragma unroll
            for (int kt = 0; kt < KT; ++kt) {
                v16bf A = load_a_frag(inBuf, pitchIn, mt, kt, hi, ln);
                acc = __builtin_amdgcn_wmma_f32_16x16x32_bf16(
                    false, A, false, Bf[kt], (short)0, acc, false, false);
            }
            store_c<RELU>(outBuf, pitchOut, mt, ntile, acc, hi, ln);
        }
    }
}

// ---------------- kernels ----------------
__global__ void zero_ws_kernel(uint4* p, long long n4) {
    long long i = (long long)blockIdx.x * blockDim.x + threadIdx.x;
    long long stride = (long long)gridDim.x * blockDim.x;
    uint4 z = {0u, 0u, 0u, 0u};
    for (; i < n4; i += stride) p[i] = z;
}

// fold BN into biases (992 floats: b0e[64] b1e[128] b2e[256] b3e[512] bce[32])
__global__ void fold_bias_kernel(const float* b0, const float* b1, const float* b2,
                                 const float* b3, const float* bc,
                                 const float* g0, const float* be0, const float* m0, const float* v0,
                                 const float* g1, const float* be1, const float* m1, const float* v1,
                                 const float* g2, const float* be2, const float* m2, const float* v2,
                                 const float* g3, const float* be3, const float* m3, const float* v3,
                                 const float* w0, float* out) {
    int t = blockIdx.x * blockDim.x + threadIdx.x;
    if (t < 64) {
        float s1 = g1[t] * rsqrtf(v1[t] + BN_EPS);
        float t1 = be1[t] - m1[t] * s1;
        float acc = b0[t];
        for (int i = 0; i < 7; ++i) {
            float s0 = g0[i] * rsqrtf(v0[i] + BN_EPS);
            float t0 = be0[i] - m0[i] * s0;
            acc += t0 * w0[t * 7 + i];
        }
        out[t] = s1 * acc + t1;
    } else if (t < 192) {
        int o = t - 64;
        float s = g2[o] * rsqrtf(v2[o] + BN_EPS);
        out[t] = s * b1[o] + (be2[o] - m2[o] * s);
    } else if (t < 448) {
        int o = t - 192;
        float s = g3[o] * rsqrtf(v3[o] + BN_EPS);
        out[t] = s * b2[o] + (be3[o] - m3[o] * s);
    } else if (t < 960) {
        out[t] = b3[t - 448];
    } else if (t < 992) {
        out[t] = bc[t - 960];
    }
}

// fold BN scales into weights, convert bf16, swizzle into B-fragment layout.
// B fragment (32x16 bf16): lane n=L&15, hi=L>>4; dword p holds K = (p>>2)*16 + hi*8 + 2*(p&3) (+1).
__global__ void fold_wfrag_kernel(const float* __restrict__ w, int KI, int KT,
                                  const float* gIn, const float* vIn,
                                  const float* gOut, const float* vOut,
                                  unsigned* __restrict__ wf, int nFrags) {
    int t = blockIdx.x * blockDim.x + threadIdx.x;
    if (t >= nFrags * 256) return;
    int frag = t >> 8, r = t & 255, p = r >> 5, lane = r & 31;
    int hi = lane >> 4, nloc = lane & 15;
    int ntile = frag / KT, kt = frag % KT;
    int n = ntile * 16 + nloc;
    float so = 1.0f;
    if (gOut) so = gOut[n] * rsqrtf(vOut[n] + BN_EPS);
    unsigned out = 0;
    for (int j = 0; j < 2; ++j) {
        int k = kt * 32 + (p >> 2) * 16 + hi * 8 + 2 * (p & 3) + j;
        float val = 0.0f;
        if (k < KI) {
            float si = 1.0f;
            if (gIn) si = gIn[k] * rsqrtf(vIn[k] + BN_EPS);
            val = w[(size_t)n * KI + k] * si * so;
        }
        out |= ((unsigned)f2bf(val)) << (16 * j);
    }
    wf[(size_t)frag * 256 + p * 32 + lane] = out;
}

// per-point voxel id + cap at MAXPT points per voxel
__global__ void rank_kernel(const int* __restrict__ xy, int* __restrict__ cnt,
                            int* __restrict__ pvox) {
    int t = blockIdx.x * blockDim.x + threadIdx.x;
    if (t >= BATCH * NPTS) return;
    int b = t / NPTS;
    int x = xy[(size_t)t * 2 + 0];
    int y = xy[(size_t)t * 2 + 1];
    int vox = x * GYY + y;
    int old = atomicAdd(&cnt[(size_t)b * NUM_VOX + vox], 1);
    pvox[t] = (old < MAXPT) ? vox : -1;
}

// fused 4-layer pointnet over a 64-point tile + segment-max scatter
__global__ void __launch_bounds__(128) mlp_kernel(const float* __restrict__ pt,
                                                  const int* __restrict__ pvox,
                                                  const unsigned* __restrict__ wf0,
                                                  const unsigned* __restrict__ wf1,
                                                  const unsigned* __restrict__ wf2,
                                                  const unsigned* __restrict__ wf3,
                                                  const float* __restrict__ biases,
                                                  unsigned* __restrict__ vmax) {
    __shared__ __align__(16) unsigned char sX[TILE_M * P256];   // 33792 B
    __shared__ __align__(16) unsigned char sY[TILE_M * P512];   // 66560 B
    __shared__ int svox[TILE_M];

    const int tid = threadIdx.x, wave = tid >> 5, lane = tid & 31;
    const int b = blockIdx.x / TILES_PER_B;
    const int tile = blockIdx.x % TILES_PER_B;
    const long n0 = (long)tile * TILE_M;

    // stage input (K padded 7 -> 32, bf16, pitched) into sY
    for (int e = tid; e < TILE_M * 32; e += 128) {
        int row = e >> 5, k = e & 31;
        long n = n0 + row;
        unsigned short h = 0;
        if (k < 7 && n < NPTS) h = f2bf(pt[((long)b * NPTS + n) * 7 + k]);
        *(unsigned short*)(sY + (size_t)row * P32 + k * 2) = h;
    }
    if (tid < TILE_M) {
        long n = n0 + tid;
        svox[tid] = (n < NPTS) ? pvox[(size_t)b * NPTS + n] : -1;
    }
    __syncthreads();

    mlp_layer<1, 1, true >(wf0, biases + 0,   sY, P32,  sX, P64,  wave, lane); __syncthreads();
    mlp_layer<2, 2, true >(wf1, biases + 64,  sX, P64,  sY, P128, wave, lane); __syncthreads();
    mlp_layer<4, 4, true >(wf2, biases + 192, sY, P128, sX, P256, wave, lane); __syncthreads();
    mlp_layer<8, 8, false>(wf3, biases + 448, sX, P256, sY, P512, wave, lane); __syncthreads();

    // segment-max scatter (order-preserving u32 encoding)
    for (int e = tid; e < TILE_M * 512; e += 128) {
        int row = e >> 9, c = e & 511;
        int vox = svox[row];
        if (vox >= 0) {
            unsigned short h = *(const unsigned short*)(sY + (size_t)row * P512 + c * 2);
            unsigned u = encf(bf2f(h));
            atomicMax(&vmax[((size_t)b * NUM_VOX + vox) * 512 + c], u);
        }
    }
}

// 512 -> 32 compression GEMM over 64 voxels/workgroup + relu + planar store
__global__ void __launch_bounds__(128) compress_kernel(const unsigned* __restrict__ vmax,
                                                       const int* __restrict__ cnt,
                                                       const unsigned* __restrict__ wfc,
                                                       const float* __restrict__ bce,
                                                       float* __restrict__ out) {
    __shared__ __align__(16) unsigned char sA[TILE_M * P512];   // 66560 B
    __shared__ float sOut[TILE_M][33];
    __shared__ int socc[TILE_M];

    const int tid = threadIdx.x, wave = tid >> 5, lane = tid & 31;
    const int b = blockIdx.x / VOX_TILES;
    const int v0 = (blockIdx.x % VOX_TILES) * TILE_M;

    for (int e = tid; e < TILE_M * 512; e += 128) {
        int row = e >> 9, k = e & 511;
        unsigned u = vmax[((size_t)b * NUM_VOX + v0 + row) * 512 + k];
        *(unsigned short*)(sA + (size_t)row * P512 + k * 2) = f2bf(decf(u));
    }
    if (tid < TILE_M) socc[tid] = cnt[(size_t)b * NUM_VOX + v0 + tid];
    __syncthreads();

    // 2 n-tiles x 4 m-tiles split over 4 waves; KT = 16
    const int ntile = wave & 1;
    const int mbase = (wave >> 1) * 2;
    const int hi = lane >> 4, ln = lane & 15;
    for (int mi = 0; mi < 2; ++mi) {
        const int mt = mbase + mi;
        const float bv = bce[ntile * 16 + ln];
        v8f acc = {bv, bv, bv, bv, bv, bv, bv, bv};
#pragma unroll
        for (int kt = 0; kt < 16; ++kt) {
            v16bf A = load_a_frag(sA, P512, mt, kt, hi, ln);
            v16bf Bf = load_b_frag(wfc, ntile * 16 + kt, lane);
            acc = __builtin_amdgcn_wmma_f32_16x16x32_bf16(
                false, A, false, Bf, (short)0, acc, false, false);
        }
#pragma unroll
        for (int r = 0; r < 8; ++r)
            sOut[mt * 16 + r + hi * 8][ntile * 16 + ln] = fmaxf(acc[r], 0.0f);
    }
    __syncthreads();

    for (int e = tid; e < 32 * TILE_M; e += 128) {
        int c = e >> 6, row = e & 63;
        float val = (socc[row] > 0) ? sOut[row][c] : 0.0f;
        out[((size_t)b * 32 + c) * NUM_VOX + v0 + row] = val;
    }
}

// ---------------- launch ----------------
extern "C" void kernel_launch(void* const* d_in, const int* in_sizes, int n_in,
                              void* d_out, int out_size, void* d_ws, size_t ws_size,
                              hipStream_t stream) {
    const float* pt_fea = (const float*)d_in[0];
    const int*   xy_ind = (const int*)d_in[1];
    const float* w0 = (const float*)d_in[2];
    const float* b0 = (const float*)d_in[3];
    const float* w1 = (const float*)d_in[4];
    const float* b1 = (const float*)d_in[5];
    const float* w2 = (const float*)d_in[6];
    const float* b2 = (const float*)d_in[7];
    const float* w3 = (const float*)d_in[8];
    const float* b3 = (const float*)d_in[9];
    const float* g0 = (const float*)d_in[10]; const float* be0 = (const float*)d_in[11];
    const float* m0 = (const float*)d_in[12]; const float* v0  = (const float*)d_in[13];
    const float* g1 = (const float*)d_in[14]; const float* be1 = (const float*)d_in[15];
    const float* m1 = (const float*)d_in[16]; const float* v1  = (const float*)d_in[17];
    const float* g2 = (const float*)d_in[18]; const float* be2 = (const float*)d_in[19];
    const float* m2 = (const float*)d_in[20]; const float* v2  = (const float*)d_in[21];
    const float* g3 = (const float*)d_in[22]; const float* be3 = (const float*)d_in[23];
    const float* m3 = (const float*)d_in[24]; const float* v3  = (const float*)d_in[25];
    const float* wc = (const float*)d_in[26];
    const float* bc = (const float*)d_in[27];

    unsigned char* ws = (unsigned char*)d_ws;
    unsigned* vmax   = (unsigned*)ws;
    int*      cnt    = (int*)(ws + CNT_OFF);
    int*      pvox   = (int*)(ws + PVOX_OFF);
    unsigned* wf0    = (unsigned*)(ws + WF0_OFF);
    unsigned* wf1    = (unsigned*)(ws + WF1_OFF);
    unsigned* wf2    = (unsigned*)(ws + WF2_OFF);
    unsigned* wf3    = (unsigned*)(ws + WF3_OFF);
    unsigned* wfc    = (unsigned*)(ws + WFC_OFF);
    float*    biases = (float*)(ws + BIAS_OFF);

    // 1) zero vmax + cnt
    zero_ws_kernel<<<8192, 256, 0, stream>>>((uint4*)ws, (long long)ZERO_U4);

    // 2) fold BN into biases + fragment-swizzled bf16 weights
    fold_bias_kernel<<<4, 256, 0, stream>>>(b0, b1, b2, b3, bc,
                                            g0, be0, m0, v0, g1, be1, m1, v1,
                                            g2, be2, m2, v2, g3, be3, m3, v3,
                                            w0, biases);
    fold_wfrag_kernel<<<NFRAG0, 256, 0, stream>>>(w0, 7,   1,  g0, v0, g1, v1, wf0, NFRAG0);
    fold_wfrag_kernel<<<NFRAG1, 256, 0, stream>>>(w1, 64,  2,  nullptr, nullptr, g2, v2, wf1, NFRAG1);
    fold_wfrag_kernel<<<NFRAG2, 256, 0, stream>>>(w2, 128, 4,  nullptr, nullptr, g3, v3, wf2, NFRAG2);
    fold_wfrag_kernel<<<NFRAG3, 256, 0, stream>>>(w3, 256, 8,  nullptr, nullptr, nullptr, nullptr, wf3, NFRAG3);
    fold_wfrag_kernel<<<NFRAGC, 256, 0, stream>>>(wc, 512, 16, nullptr, nullptr, nullptr, nullptr, wfc, NFRAGC);

    // 3) voxel ids + per-voxel point cap
    rank_kernel<<<(BATCH * NPTS + 255) / 256, 256, 0, stream>>>(xy_ind, cnt, pvox);

    // 4) fused pointnet MLP (bf16 WMMA) + segment-max scatter
    mlp_kernel<<<BATCH * TILES_PER_B, 128, 0, stream>>>(pt_fea, pvox, wf0, wf1, wf2, wf3,
                                                        biases, vmax);

    // 5) 512->32 compression (bf16 WMMA) + relu + planar output
    compress_kernel<<<BATCH * VOX_TILES, 128, 0, stream>>>(vmax, cnt, wfc, biases + 960,
                                                           (float*)d_out);
}
